// AttentionSACModel_8993661518717
// MI455X (gfx1250) — matrix operand order repaired and checked by reading, not compiled, not run on hardware
//
#include <hip/hip_runtime.h>
#include <hip/hip_bf16.h>

#define OWN_DIM 7
#define INTR_DIM 5
#define N_HEADS 3
#define HEAD_DIM 5
#define N_INTR 256
#define ACT_DIM 2
#define HID 256

#define ROWS_PER_BLOCK 64
#define THREADS 128
#define K1PAD 32
#define OBS_DIM (OWN_DIM + N_INTR * INTR_DIM)  // 1287

typedef __attribute__((ext_vector_type(16))) _Float16 v16h;
typedef __attribute__((ext_vector_type(8)))  float    v8f;

__device__ __forceinline__ float fast_tanh(float x) {
    x = fminf(fmaxf(x, -15.f), 15.f);
    float t = __expf(2.f * x);
    return (t - 1.f) / (t + 1.f);
}

// A fragment (16x32 f16): lanes 0-15 hold row m=lane, K {0..15};
// lanes 16-31 hold row m=lane-16, K {16..31}.
__device__ __forceinline__ v16h load_a_frag(const _Float16* rowp, int halfw) {
    v16h a;
    const int o0 = halfw * 16;
    const int o1 = halfw * 16 + 8;
#pragma unroll
    for (int i = 0; i < 8; ++i) { a[i] = rowp[o0 + i]; a[8 + i] = rowp[o1 + i]; }
    return a;
}

// B fragment (32x16 f16), weights stored n-major with contiguous K:
// lane holds 16 contiguous K values for column n = lane&15, K-half = lane>>4.
__device__ __forceinline__ v16h load_b_frag(const _Float16* p) {
    v16h b;
#pragma unroll
    for (int i = 0; i < 16; ++i) b[i] = p[i];
    return b;
}

// ---- prep: convert W1 (22x256) -> f16 [256][32] (n-major, zero K-pad),
//            convert W2 (256x256) -> f16 [256][256] (n-major) ----
__global__ void prep_weights_kernel(const float* __restrict__ W1,
                                    const float* __restrict__ W2,
                                    _Float16* __restrict__ W1t,
                                    _Float16* __restrict__ W2t) {
    int idx = blockIdx.x * blockDim.x + threadIdx.x;
    const int n1 = HID * K1PAD;          // 8192
    const int n2 = HID * HID;            // 65536
    if (idx < n1) {
        int n = idx >> 5, k = idx & 31;
        W1t[idx] = (k < (OWN_DIM + N_HEADS * HEAD_DIM))
                       ? (_Float16)W1[k * HID + n] : (_Float16)0.f;
    } else if (idx < n1 + n2) {
        int j = idx - n1;
        int n = j >> 8, k = j & 255;
        W2t[j] = (_Float16)W2[k * HID + n];
    }
}

__global__ __launch_bounds__(THREADS) void attn_mlp_kernel(
    const float* __restrict__ obs,
    const float* __restrict__ Wq, const float* __restrict__ bq,
    const float* __restrict__ Wk, const float* __restrict__ bk,
    const float* __restrict__ Wv, const float* __restrict__ bv,
    const float* __restrict__ v_att, const float* __restrict__ temp,
    const float* __restrict__ b1, const float* __restrict__ b2,
    const float* __restrict__ Wf, const float* __restrict__ bf,
    const float* __restrict__ log_std,
    const _Float16* __restrict__ W1t, const _Float16* __restrict__ W2t,
    float* __restrict__ out)
{
    __shared__ _Float16 sX[ROWS_PER_BLOCK * K1PAD];   // 4 KB
    __shared__ _Float16 sH1[ROWS_PER_BLOCK * HID];    // 32 KB

    const int t       = threadIdx.x;
    const int lane    = t & 31;
    const int wave    = t >> 5;
    const int rowLoc  = t >> 1;          // 0..63 : two threads per row
    const int half128 = t & 1;           // which half of the 256 interactions
    const float T = fabsf(temp[0]);

    // ================= Phase 1: per-row attention (online softmax) =========
    const float* orow = obs + (long)(blockIdx.x * ROWS_PER_BLOCK + rowLoc) * OBS_DIM;
    float own[OWN_DIM];
#pragma unroll
    for (int i = 0; i < OWN_DIM; ++i) own[i] = orow[i];

    float ctx_all[N_HEADS * HEAD_DIM];

    for (int h = 0; h < N_HEADS; ++h) {
        const float* wqh = Wq + h * HEAD_DIM * OWN_DIM;
        const float* wkh = Wk + h * HEAD_DIM * INTR_DIM;
        const float* wvh = Wv + h * HEAD_DIM * INTR_DIM;
        float kw[HEAD_DIM * INTR_DIM], vw[HEAD_DIM * INTR_DIM];
#pragma unroll
        for (int i = 0; i < HEAD_DIM * INTR_DIM; ++i) { kw[i] = wkh[i]; vw[i] = wvh[i]; }
        float q[HEAD_DIM], bkr[HEAD_DIM], bvr[HEAD_DIM], va[HEAD_DIM];
#pragma unroll
        for (int d = 0; d < HEAD_DIM; ++d) {
            float acc = bq[h * HEAD_DIM + d];
#pragma unroll
            for (int o = 0; o < OWN_DIM; ++o) acc = fmaf(own[o], wqh[d * OWN_DIM + o], acc);
            q[d]  = acc;
            bkr[d] = bk[h * HEAD_DIM + d];
            bvr[d] = bv[h * HEAD_DIM + d];
            va[d]  = v_att[h * HEAD_DIM + d];
        }

        float m = -__builtin_inff(), s = 0.f;
        float c[HEAD_DIM] = {0.f, 0.f, 0.f, 0.f, 0.f};
        const float* ip = orow + OWN_DIM + half128 * (128 * INTR_DIM);
        for (int nc = 0; nc < 16; ++nc) {
            // pull the stream ~2 chunks (320B) ahead into the WGP cache
            __builtin_prefetch(ip + 80, 0, 1);
#pragma unroll
            for (int nn = 0; nn < 8; ++nn) {
                float x0 = ip[0], x1 = ip[1], x2 = ip[2], x3 = ip[3], x4 = ip[4];
                ip += INTR_DIM;
                float sa = fabsf(x0) + fabsf(x1) + fabsf(x2) + fabsf(x3) + fabsf(x4);
                if (sa >= 1e-6f) {                // padding slot -> weight 0
                    float e = 0.f;
                    float vv[HEAD_DIM];
#pragma unroll
                    for (int d = 0; d < HEAD_DIM; ++d) {
                        const float* kr = kw + d * INTR_DIM;
                        const float* vr = vw + d * INTR_DIM;
                        float kk = bkr[d];
                        kk = fmaf(kr[0], x0, kk); kk = fmaf(kr[1], x1, kk);
                        kk = fmaf(kr[2], x2, kk); kk = fmaf(kr[3], x3, kk);
                        kk = fmaf(kr[4], x4, kk);
                        float vd = bvr[d];
                        vd = fmaf(vr[0], x0, vd); vd = fmaf(vr[1], x1, vd);
                        vd = fmaf(vr[2], x2, vd); vd = fmaf(vr[3], x3, vd);
                        vd = fmaf(vr[4], x4, vd);
                        vv[d] = vd;
                        e = fmaf(fast_tanh(q[d] + kk), va[d], e);
                    }
                    float score = e * T;
                    if (score > m) {
                        float sc = __expf(m - score);  // exp(-inf)=0 first hit
                        s = fmaf(s, sc, 1.f);
#pragma unroll
                        for (int d = 0; d < HEAD_DIM; ++d) c[d] = fmaf(c[d], sc, vv[d]);
                        m = score;
                    } else {
                        float w = __expf(score - m);
                        s += w;
#pragma unroll
                        for (int d = 0; d < HEAD_DIM; ++d) c[d] = fmaf(w, vv[d], c[d]);
                    }
                }
            }
        }
        // merge the two half-row partials (adjacent lanes)
        float m2 = __shfl_xor(m, 1, 32);
        float s2 = __shfl_xor(s, 1, 32);
        float M  = fmaxf(m, m2);
        float w1 = (s  > 0.f) ? __expf(m  - M) : 0.f;
        float w2 = (s2 > 0.f) ? __expf(m2 - M) : 0.f;
        float S  = s * w1 + s2 * w2;
        float inv = (S > 0.f) ? 1.f / S : 0.f;   // nan_to_num path
#pragma unroll
        for (int d = 0; d < HEAD_DIM; ++d) {
            float c2 = __shfl_xor(c[d], 1, 32);
            ctx_all[h * HEAD_DIM + d] = (c[d] * w1 + c2 * w2) * inv;
        }
    }

    if (half128 == 0) {
        _Float16* xr = sX + rowLoc * K1PAD;
#pragma unroll
        for (int i = 0; i < OWN_DIM; ++i) xr[i] = (_Float16)own[i];
#pragma unroll
        for (int d = 0; d < N_HEADS * HEAD_DIM; ++d) xr[OWN_DIM + d] = (_Float16)ctx_all[d];
#pragma unroll
        for (int j = OWN_DIM + N_HEADS * HEAD_DIM; j < K1PAD; ++j) xr[j] = (_Float16)0.f;
    }
    __syncthreads();

    // ================= Phase 2: WMMA MLP (one 16-row tile per wave) =========
    const int wrow0  = wave * 16;        // tile base row within block
    const int nlocal = lane & 15;        // also the A-fragment row index
    const int halfw  = lane >> 4;

    // ---- layer 1: x(16x32) @ W1t -> h1 (leaky relu), f16 into LDS ----
    {
        v16h a1 = load_a_frag(sX + (wrow0 + nlocal) * K1PAD, halfw);
#pragma unroll
        for (int nb = 0; nb < HID / 16; ++nb) {
            int col = nb * 16 + nlocal;
            v16h bfr = load_b_frag(W1t + col * K1PAD + halfw * 16);
            v8f acc = {};
            acc = __builtin_amdgcn_wmma_f32_16x16x32_f16(
                false, a1, false, bfr, (short)0, acc, false, false);
            float bias = b1[col];
#pragma unroll
            for (int r = 0; r < 8; ++r) {
                int mrow = r + halfw * 8;
                float y = acc[r] + bias;
                y = fmaxf(y, 0.2f * y);          // leaky relu
                sH1[(wrow0 + mrow) * HID + col] = (_Float16)y;
            }
        }
    }
    __syncthreads();

    // ---- layer 2 (h1 @ W2, K=256) fused with final 256->2 projection ----
    // A fragments depend only on the K-chunk: hoist all 8 into registers so
    // each is read from LDS once instead of once per output block (16x less
    // LDS traffic, the shared resource across the WGP's waves).
    v16h a2[HID / 32];
#pragma unroll
    for (int kc = 0; kc < HID / 32; ++kc)
        a2[kc] = load_a_frag(sH1 + (wrow0 + nlocal) * HID + kc * 32, halfw);

    float po0[8], po1[8];
#pragma unroll
    for (int r = 0; r < 8; ++r) { po0[r] = 0.f; po1[r] = 0.f; }

    for (int nb = 0; nb < HID / 16; ++nb) {
        v8f acc = {};
#pragma unroll
        for (int kc = 0; kc < HID / 32; ++kc) {
            v16h b = load_b_frag(W2t + (nb * 16 + nlocal) * HID + kc * 32 + halfw * 16);
            acc = __builtin_amdgcn_wmma_f32_16x16x32_f16(
                false, a2[kc], false, b, (short)0, acc, false, false);
        }
        int col = nb * 16 + nlocal;
        float bias = b2[col];
        float wf0 = Wf[col * ACT_DIM + 0];
        float wf1 = Wf[col * ACT_DIM + 1];
#pragma unroll
        for (int r = 0; r < 8; ++r) {
            float y = acc[r] + bias;
            y = fmaxf(y, 0.2f * y);
            po0[r] = fmaf(y, wf0, po0[r]);
            po1[r] = fmaf(y, wf1, po1[r]);
        }
    }

    // butterfly reduce across the 16 lanes sharing each C-tile row group
#pragma unroll
    for (int r = 0; r < 8; ++r) {
#pragma unroll
        for (int mask = 1; mask < 16; mask <<= 1) {
            po0[r] += __shfl_xor(po0[r], mask, 32);
            po1[r] += __shfl_xor(po1[r], mask, 32);
        }
    }

    if (nlocal == 0) {
        float bf0 = bf[0], bf1 = bf[1];
        float ls0 = log_std[0], ls1 = log_std[1];
#pragma unroll
        for (int r = 0; r < 8; ++r) {
            int grow = blockIdx.x * ROWS_PER_BLOCK + wrow0 + halfw * 8 + r;
            float4 o = make_float4(po0[r] + bf0, po1[r] + bf1, ls0, ls1);
            *(float4*)(out + grow * 4) = o;
        }
    }
}

extern "C" void kernel_launch(void* const* d_in, const int* in_sizes, int n_in,
                              void* d_out, int out_size, void* d_ws, size_t ws_size,
                              hipStream_t stream) {
    const float* obs     = (const float*)d_in[0];
    const float* Wq      = (const float*)d_in[1];
    const float* bq      = (const float*)d_in[2];
    const float* Wk      = (const float*)d_in[3];
    const float* bk      = (const float*)d_in[4];
    const float* Wv      = (const float*)d_in[5];
    const float* bv      = (const float*)d_in[6];
    const float* v_att   = (const float*)d_in[7];
    const float* temp    = (const float*)d_in[8];
    const float* W1      = (const float*)d_in[9];
    const float* b1      = (const float*)d_in[10];
    const float* W2      = (const float*)d_in[11];
    const float* b2      = (const float*)d_in[12];
    const float* Wf      = (const float*)d_in[13];
    const float* bf      = (const float*)d_in[14];
    const float* log_std = (const float*)d_in[15];
    float* out = (float*)d_out;

    _Float16* W1t = (_Float16*)d_ws;               // 256*32  f16 = 16 KB
    _Float16* W2t = W1t + HID * K1PAD;             // 256*256 f16 = 128 KB

    const int prep_total = HID * K1PAD + HID * HID;
    prep_weights_kernel<<<(prep_total + 255) / 256, 256, 0, stream>>>(W1, W2, W1t, W2t);

    const int B = in_sizes[0] / OBS_DIM;           // 16384
    attn_mlp_kernel<<<B / ROWS_PER_BLOCK, THREADS, 0, stream>>>(
        obs, Wq, bq, Wk, bk, Wv, bv, v_att, temp,
        b1, b2, Wf, bf, log_std, W1t, W2t, out);
}
